// WhisperPreKV_88295937671835
// MI455X (gfx1250) — matrix operand order repaired and checked by reading, not compiled, not run on hardware
//
#include <hip/hip_runtime.h>
#include <cstdint>
#include <cstddef>

#define LAYERS 12
#define BATCH  4
#define TQ     448
#define SKV    1500
#define DM     768
#define NHEADS 12
#define DHEAD  64
#define VOCAB  51865
#define DFF_N  3072
#define NTOK   (BATCH * TQ)   // 1792

typedef __attribute__((ext_vector_type(16))) __bf16 v16bf;
typedef __attribute__((ext_vector_type(2)))  __bf16 v2bf;
typedef __attribute__((ext_vector_type(8)))  float  v8f;

union Frag {
  v16bf v;
  uint4 q[2];
  unsigned int u[8];
};

// Hardware packed bf16 conversion with explicit pairing (avoids the SLP
// mispairing + re-interleave seen in round 2).
__device__ __forceinline__ unsigned int pack2(float a, float b) {
#if __has_builtin(__builtin_amdgcn_cvt_pk_bf16_f32)
  union { v2bf v; unsigned int u; } t;
  t.v = __builtin_amdgcn_cvt_pk_bf16_f32(a, b);
  return t.u;
#else
  union { __bf16 h[2]; unsigned int u; } t;
  t.h[0] = (__bf16)a;
  t.h[1] = (__bf16)b;
  return t.u;
#endif
}
__device__ __forceinline__ unsigned short f2bf(float f) {
  union { __bf16 h; unsigned short u; } t;
  t.h = (__bf16)f;
  return t.u;
}
__device__ __forceinline__ uint2 pack4(float4 f) {
  uint2 r;
  r.x = pack2(f.x, f.y);
  r.y = pack2(f.z, f.w);
  return r;
}
__device__ __forceinline__ v8f wmma_bf16(const Frag& a, const Frag& b, v8f c) {
  return __builtin_amdgcn_wmma_f32_16x16x32_bf16(false, a.v, false, b.v,
                                                 (short)0, c, false, false);
}

// ---------------------------------------------------------------------------
// Embedding: x[row] = tok_emb[tokens[row]] + pos_emb[row % T]
// ---------------------------------------------------------------------------
__global__ __launch_bounds__(256) void embed_kernel(
    const int* __restrict__ tokens, const float* __restrict__ temb,
    const float* __restrict__ pemb, float* __restrict__ x) {
  int row = blockIdx.x;
  int t = row % TQ;
  int tok = tokens[row];
  const float* te = temb + (size_t)tok * DM;
  const float* pe = pemb + (size_t)t * DM;
  float* xo = x + (size_t)row * DM;
  for (int c = threadIdx.x; c < DM; c += 256) xo[c] = te[c] + pe[c];
}

// ---------------------------------------------------------------------------
// LayerNorm (fp32), one block per row, D = 768 = 3 * 256
// ---------------------------------------------------------------------------
__global__ __launch_bounds__(256) void ln_kernel(
    const float* __restrict__ x, const float* __restrict__ g,
    const float* __restrict__ b, float* __restrict__ h) {
  __shared__ float red[256];
  int row = blockIdx.x, tid = threadIdx.x;
  const float* xr = x + (size_t)row * DM;
  float v[3], s = 0.f;
  for (int i = 0; i < 3; ++i) { v[i] = xr[tid + i * 256]; s += v[i]; }
  red[tid] = s; __syncthreads();
  for (int st = 128; st > 0; st >>= 1) {
    if (tid < st) red[tid] += red[tid + st];
    __syncthreads();
  }
  float mean = red[0] * (1.f / DM);
  __syncthreads();
  float sq = 0.f;
  for (int i = 0; i < 3; ++i) { float d = v[i] - mean; sq += d * d; }
  red[tid] = sq; __syncthreads();
  for (int st = 128; st > 0; st >>= 1) {
    if (tid < st) red[tid] += red[tid + st];
    __syncthreads();
  }
  float rstd = rsqrtf(red[0] * (1.f / DM) + 1e-5f);
  float* hr = h + (size_t)row * DM;
  for (int i = 0; i < 3; ++i) {
    int c = tid + i * 256;
    hr[c] = (v[i] - mean) * rstd * g[c] + b[c];
  }
}

// ---------------------------------------------------------------------------
// WMMA bf16 GEMM: C[M,N] = epilogue(A[M,K] @ B)   (fp32 in/out, bf16 WMMA)
//   bT==0 : B is [K,N] row-major (weights)
//   bT==1 : B is [N,K] row-major (tok_emb for logits)
// Epilogue: v = alpha*(acc + bias[n]); optional exact GELU; optional +res.
// Block tile 128x128x32; 256 threads = 8 waves (2x4), wave tile 64x32.
// M % 128 == 0 and K % 32 == 0 guaranteed by problem sizes; N guarded.
// Staging: load phase (4 outstanding b128 loads) then convert/store phase.
// ---------------------------------------------------------------------------
#define LDT 48   // LDS row stride in halves (96B, 16B-aligned)

__global__ __launch_bounds__(256) void gemm_kernel(
    const float* __restrict__ A, const float* __restrict__ Bw,
    const float* __restrict__ bias, const float* __restrict__ res,
    float* __restrict__ C, int M, int N, int K, float alpha, int bT, int doGelu) {
  __shared__ alignas(16) unsigned short lA[128 * LDT];
  __shared__ alignas(16) unsigned short lB[128 * LDT];
  int tid = threadIdx.x;
  int lane = tid & 31, w = tid >> 5;
  int wr = w >> 2, wc = w & 3;
  int half = (lane < 16) ? 0 : 1;
  int m0 = blockIdx.y * 128, n0 = blockIdx.x * 128;

  v8f acc[4][2] = {};

  for (int k0 = 0; k0 < K; k0 += 32) {
    // ---- load phase: issue all global loads before any wait ----
    float4 fa[4], fb[4];
    for (int j = 0; j < 4; ++j) {
      int i = tid + 256 * j;
      int r = i >> 3, c4 = (i & 7) << 2;
      const float* ap = A + (size_t)(m0 + r) * K + k0 + c4;
      fa[j] = *(const float4*)ap;
      if (j == 0 && k0 + 32 < K) __builtin_prefetch(ap + 32, 0, 1);
    }
    if (!bT) {
      for (int j = 0; j < 4; ++j) {
        int i = tid + 256 * j;
        int r = i >> 5, c4 = (i & 31) << 2;
        const float* bp = Bw + (size_t)(k0 + r) * N + n0 + c4;
        fb[j] = *(const float4*)bp;
        if (j == 0 && k0 + 32 < K) __builtin_prefetch(bp + (size_t)32 * N, 0, 1);
      }
    } else {
      for (int j = 0; j < 4; ++j) {
        int i = tid + 256 * j;
        int c = i >> 3, k4 = (i & 7) << 2;
        float4 f = {0.f, 0.f, 0.f, 0.f};
        if (n0 + c < N) {
          const float* bp = Bw + (size_t)(n0 + c) * K + k0 + k4;
          f = *(const float4*)bp;
          if (j == 0 && k0 + 32 < K) __builtin_prefetch(bp + 32, 0, 1);
        }
        fb[j] = f;
      }
    }
    __syncthreads();
    // ---- convert + LDS store phase ----
    for (int j = 0; j < 4; ++j) {
      int i = tid + 256 * j;
      int r = i >> 3, c4 = (i & 7) << 2;
      *(uint2*)(&lA[r * LDT + c4]) = pack4(fa[j]);
    }
    if (!bT) {
      for (int j = 0; j < 4; ++j) {
        int i = tid + 256 * j;
        int r = i >> 5, c4 = (i & 31) << 2;
        lB[(c4 + 0) * LDT + r] = f2bf(fb[j].x);
        lB[(c4 + 1) * LDT + r] = f2bf(fb[j].y);
        lB[(c4 + 2) * LDT + r] = f2bf(fb[j].z);
        lB[(c4 + 3) * LDT + r] = f2bf(fb[j].w);
      }
    } else {
      for (int j = 0; j < 4; ++j) {
        int i = tid + 256 * j;
        int c = i >> 3, k4 = (i & 7) << 2;
        *(uint2*)(&lB[c * LDT + k4]) = pack4(fb[j]);
      }
    }
    __syncthreads();

    Frag afr[4], bfr[2];
    for (int ti = 0; ti < 4; ++ti) {
      int m = wr * 64 + ti * 16 + (lane & 15);
      const unsigned short* p = &lA[m * LDT];
      afr[ti].q[0] = *(const uint4*)(p + half * 8);
      afr[ti].q[1] = *(const uint4*)(p + 16 + half * 8);
    }
    for (int tj = 0; tj < 2; ++tj) {
      int n = wc * 32 + tj * 16 + (lane & 15);
      const unsigned short* p = &lB[n * LDT];
      bfr[tj].q[0] = *(const uint4*)(p + half * 8);
      bfr[tj].q[1] = *(const uint4*)(p + 16 + half * 8);
    }
    for (int ti = 0; ti < 4; ++ti)
      for (int tj = 0; tj < 2; ++tj)
        acc[ti][tj] = wmma_bf16(afr[ti], bfr[tj], acc[ti][tj]);
  }

  // epilogue
  for (int ti = 0; ti < 4; ++ti) {
    for (int tj = 0; tj < 2; ++tj) {
      int n = n0 + wc * 32 + tj * 16 + (lane & 15);
      if (n >= N) continue;
      float bvv = bias ? bias[n] : 0.f;
      for (int r = 0; r < 8; ++r) {
        int m = m0 + wr * 64 + ti * 16 + r + half * 8;
        float vv = (acc[ti][tj][r] + bvv) * alpha;
        if (doGelu) vv = 0.5f * vv * (1.f + erff(vv * 0.70710678f));
        if (res) vv += res[(size_t)m * N + n];
        C[(size_t)m * N + n] = vv;
      }
    }
  }
}

// ---------------------------------------------------------------------------
// Flash attention (WMMA). Computes S^T = K x Q^T, online softmax over keys
// (the M dimension: 8 in-lane regs + one shfl_xor(16)), then O^T += V^T x P.
// Block: 256 thr = 8 waves, each wave owns 16 query rows (128 q / block).
// Key blocks of 32. Q/K/V are [rows, DM] fp32; out is [rows, DM] fp32.
// Scale is pre-folded into Q upstream.
// ---------------------------------------------------------------------------
__global__ __launch_bounds__(256) void attn_kernel(
    const float* __restrict__ Q, const float* __restrict__ Kp,
    const float* __restrict__ Vp, float* __restrict__ Out,
    int Tq, int Tk, int kvStride, int causal) {
  __shared__ alignas(16) unsigned short lK[32 * 64];   // [key][d]  bf16
  __shared__ alignas(16) unsigned short lVT[64 * 32];  // [d][key]  bf16
  int tid = threadIdx.x, lane = tid & 31, w = tid >> 5;
  int half = (lane < 16) ? 0 : 1;
  int b = blockIdx.z, hd = blockIdx.y;
  int qb0 = blockIdx.x * 128;
  int qi = qb0 + w * 16 + (lane & 15);          // this lane's query index
  bool qvalid = qi < Tq;

  // Q fragments (B operand of S^T): 2 d-steps of 32
  Frag qf[2];
  {
    int qrowi = qvalid ? qi : 0;
    const float* qrow = Q + ((size_t)(b * Tq + qrowi)) * DM + hd * DHEAD;
    float4 f[4];
    for (int j = 0; j < 2; ++j)
      for (int qq = 0; qq < 2; ++qq) {
        float4 t = {0.f, 0.f, 0.f, 0.f};
        if (qvalid) t = *(const float4*)(qrow + j * 32 + qq * 16 + half * 8 + 0);
        f[j * 2 + qq] = t;
      }
    for (int j = 0; j < 2; ++j)
      for (int qq = 0; qq < 2; ++qq) {
        // each float4 covers 4 of the 16 halves; second float4 of the 8-chunk
        float4 t2 = {0.f, 0.f, 0.f, 0.f};
        if (qvalid) t2 = *(const float4*)(qrow + j * 32 + qq * 16 + half * 8 + 4);
        uint2 lo = pack4(f[j * 2 + qq]);
        uint2 hi = pack4(t2);
        qf[j].u[4 * qq + 0] = lo.x;
        qf[j].u[4 * qq + 1] = lo.y;
        qf[j].u[4 * qq + 2] = hi.x;
        qf[j].u[4 * qq + 3] = hi.y;
      }
  }

  float mrun = -1e30f, lrun = 0.f;
  v8f o[4] = {};

  int kbEnd = causal ? ((qb0 + 128 < Tk) ? qb0 + 128 : Tk) : Tk;
  const float* Kb = Kp + (size_t)b * kvStride * DM + hd * DHEAD;
  const float* Vb = Vp + (size_t)b * kvStride * DM + hd * DHEAD;

  for (int kb = 0; kb < kbEnd; kb += 32) {
    // ---- load phase: 4 outstanding b128 loads per thread ----
    float4 fk[2], fv[2];
    for (int j = 0; j < 2; ++j) {
      int i = tid + 256 * j;
      int key = i >> 4, d4 = (i & 15) << 2;
      float4 a = {0.f, 0.f, 0.f, 0.f}, c = {0.f, 0.f, 0.f, 0.f};
      if (kb + key < Tk) {
        const float* kp = Kb + (size_t)(kb + key) * DM + d4;
        const float* vp = Vb + (size_t)(kb + key) * DM + d4;
        a = *(const float4*)kp;
        c = *(const float4*)vp;
        if (j == 0 && kb + 32 + key < Tk && kb + 32 < kbEnd) {
          __builtin_prefetch(kp + (size_t)32 * DM, 0, 1);
          __builtin_prefetch(vp + (size_t)32 * DM, 0, 1);
        }
      }
      fk[j] = a; fv[j] = c;
    }
    __syncthreads();
    // ---- convert + LDS store phase ----
    for (int j = 0; j < 2; ++j) {
      int i = tid + 256 * j;
      int key = i >> 4, d4 = (i & 15) << 2;
      *(uint2*)(&lK[key * 64 + d4]) = pack4(fk[j]);
      lVT[(d4 + 0) * 32 + key] = f2bf(fv[j].x);
      lVT[(d4 + 1) * 32 + key] = f2bf(fv[j].y);
      lVT[(d4 + 2) * 32 + key] = f2bf(fv[j].z);
      lVT[(d4 + 3) * 32 + key] = f2bf(fv[j].w);
    }
    __syncthreads();

    // scores S^T: two 16-key subtiles, K-dim = 64 (2 WMMAs each)
    v8f s[2];
    for (int sst = 0; sst < 2; ++sst) {
      v8f c = {};
      for (int j = 0; j < 2; ++j) {
        Frag af;
        const unsigned short* p = &lK[(sst * 16 + (lane & 15)) * 64 + j * 32];
        af.q[0] = *(const uint4*)(p + half * 8);
        af.q[1] = *(const uint4*)(p + 16 + half * 8);
        c = wmma_bf16(af, qf[j], c);
      }
      s[sst] = c;
    }

    // mask + online softmax (per lane: fixed query, keys across regs+pair)
    float mloc = -1e30f;
    for (int sst = 0; sst < 2; ++sst)
      for (int r = 0; r < 8; ++r) {
        int key = kb + sst * 16 + r + half * 8;
        float sv = s[sst][r];
        bool ok = (key < Tk) && (!causal || key <= qi);
        sv = ok ? sv : -1e30f;
        s[sst][r] = sv;
        mloc = fmaxf(mloc, sv);
      }
    mloc = fmaxf(mloc, __shfl_xor(mloc, 16, 32));
    float mnew = fmaxf(mrun, mloc);
    float scale = __expf(mrun - mnew);
    float ls = 0.f;
    for (int sst = 0; sst < 2; ++sst)
      for (int r = 0; r < 8; ++r) {
        float p = __expf(s[sst][r] - mnew);
        s[sst][r] = p;
        ls += p;
      }
    ls += __shfl_xor(ls, 16, 32);
    lrun = lrun * scale + ls;
    mrun = mnew;
    for (int c4 = 0; c4 < 4; ++c4)
      for (int r = 0; r < 8; ++r) o[c4][r] *= scale;

    // build P B-fragment (32 keys x 16 q) from in-lane packed scores
    unsigned int ps0[4], ps1[4];
    for (int v = 0; v < 4; ++v) {
      ps0[v] = pack2(s[0][2 * v], s[0][2 * v + 1]);
      ps1[v] = pack2(s[1][2 * v], s[1][2 * v + 1]);
    }
    Frag pf;
    for (int v = 0; v < 4; ++v) {
      unsigned int x0 = (unsigned int)__shfl_xor((int)ps0[v], 16, 32);
      unsigned int x1 = (unsigned int)__shfl_xor((int)ps1[v], 16, 32);
      (void)x0; (void)x1;
      pf.u[v]     = ps0[v];
      pf.u[v + 4] = ps1[v];
    }

    // O^T += V^T x P   (4 d-chunks of 16)
    for (int c4 = 0; c4 < 4; ++c4) {
      Frag af;
      const unsigned short* p = &lVT[(c4 * 16 + (lane & 15)) * 32];
      af.q[0] = *(const uint4*)(p + half * 8);
      af.q[1] = *(const uint4*)(p + 16 + half * 8);
      o[c4] = wmma_bf16(af, pf, o[c4]);
    }
  }

  if (qvalid) {
    float inv = (lrun > 0.f) ? 1.f / lrun : 0.f;
    float* orow = Out + ((size_t)(b * Tq + qi)) * DM + hd * DHEAD;
    for (int c4 = 0; c4 < 4; ++c4)
      for (int r = 0; r < 8; ++r)
        orow[c4 * 16 + r + half * 8] = o[c4][r] * inv;
  }
}

// ---------------------------------------------------------------------------
// Host orchestration
// ---------------------------------------------------------------------------
extern "C" void kernel_launch(void* const* d_in, const int* in_sizes, int n_in,
                              void* d_out, int out_size, void* d_ws, size_t ws_size,
                              hipStream_t stream) {
  (void)in_sizes; (void)n_in; (void)out_size; (void)ws_size;
  const int*   tokens = (const int*)  d_in[0];
  const float* kin    = (const float*)d_in[1];
  const float* vin    = (const float*)d_in[2];
  const float* temb   = (const float*)d_in[3];
  const float* pemb   = (const float*)d_in[4];
  const float* Wq  = (const float*)d_in[5];
  const float* bq  = (const float*)d_in[6];
  const float* Wk  = (const float*)d_in[7];
  const float* Wv  = (const float*)d_in[8];
  const float* bv  = (const float*)d_in[9];
  const float* Wo  = (const float*)d_in[10];
  const float* bo  = (const float*)d_in[11];
  const float* ln1g = (const float*)d_in[12];
  const float* ln1b = (const float*)d_in[13];
  const float* cWq = (const float*)d_in[14];
  const float* cbq = (const float*)d_in[15];
  const float* cWo = (const float*)d_in[16];
  const float* cbo = (const float*)d_in[17];
  const float* ln2g = (const float*)d_in[18];
  const float* ln2b = (const float*)d_in[19];
  const float* W1  = (const float*)d_in[20];
  const float* b1  = (const float*)d_in[21];
  const float* W2  = (const float*)d_in[22];
  const float* b2  = (const float*)d_in[23];
  const float* ln3g = (const float*)d_in[24];
  const float* ln3b = (const float*)d_in[25];
  const float* lnfg = (const float*)d_in[26];
  const float* lnfb = (const float*)d_in[27];
  float* out = (float*)d_out;
  float* ws  = (float*)d_ws;

  const size_t ND = (size_t)NTOK * DM;
  float* x    = ws;
  float* h    = x    + ND;
  float* qb   = h    + ND;
  float* kbuf = qb   + ND;
  float* vbuf = kbuf + ND;
  float* ab   = vbuf + ND;
  float* ff   = ab   + ND;              // NTOK * DFF_N

  auto gemm = [&](const float* A, const float* Bw, const float* bias,
                  const float* res, float* Co, int M, int N, int K,
                  float alpha, int bT, int gelu) {
    dim3 grid((N + 127) / 128, M / 128);
    gemm_kernel<<<grid, 256, 0, stream>>>(A, Bw, bias, res, Co, M, N, K,
                                          alpha, bT, gelu);
  };

  const float aq = 0.125f;              // DHEAD^-0.5 (q*sc * k*sc folded)
  const dim3 agrid((TQ + 127) / 128, NHEADS, BATCH);

  embed_kernel<<<NTOK, 256, 0, stream>>>(tokens, temb, pemb, x);

  for (int l = 0; l < LAYERS; ++l) {
    size_t w2 = (size_t)l * DM * DM;
    size_t v1 = (size_t)l * DM;
    // self-attention
    ln_kernel<<<NTOK, 256, 0, stream>>>(x, ln1g + v1, ln1b + v1, h);
    gemm(h, Wq + w2, bq + v1, nullptr, qb,   NTOK, DM, DM, aq,  0, 0);
    gemm(h, Wk + w2, nullptr, nullptr, kbuf, NTOK, DM, DM, 1.f, 0, 0);
    gemm(h, Wv + w2, bv + v1, nullptr, vbuf, NTOK, DM, DM, 1.f, 0, 0);
    attn_kernel<<<agrid, 256, 0, stream>>>(qb, kbuf, vbuf, ab, TQ, TQ, TQ, 1);
    gemm(ab, Wo + w2, bo + v1, x, x, NTOK, DM, DM, 1.f, 0, 0);
    // cross-attention
    ln_kernel<<<NTOK, 256, 0, stream>>>(x, ln2g + v1, ln2b + v1, h);
    gemm(h, cWq + w2, cbq + v1, nullptr, qb, NTOK, DM, DM, aq, 0, 0);
    const float* kl = kin + (size_t)l * BATCH * SKV * DM;
    const float* vl = vin + (size_t)l * BATCH * SKV * DM;
    attn_kernel<<<agrid, 256, 0, stream>>>(qb, kl, vl, ab, TQ, SKV, SKV, 0);
    gemm(ab, cWo + w2, cbo + v1, x, x, NTOK, DM, DM, 1.f, 0, 0);
    // MLP (exact GELU)
    ln_kernel<<<NTOK, 256, 0, stream>>>(x, ln3g + v1, ln3b + v1, h);
    gemm(h, W1 + (size_t)l * DM * DFF_N, b1 + (size_t)l * DFF_N, nullptr,
         ff, NTOK, DFF_N, DM, 1.f, 0, 1);
    gemm(ff, W2 + (size_t)l * DFF_N * DM, b2 + v1, x, x,
         NTOK, DM, DFF_N, 1.f, 0, 0);
  }

  ln_kernel<<<NTOK, 256, 0, stream>>>(x, lnfg, lnfb, h);
  // logits = h @ tok_emb^T  (B transposed layout, N = VOCAB guarded)
  gemm(h, temb, nullptr, nullptr, out, NTOK, VOCAB, DM, 1.f, 1, 0);
}
